// MaskedSliceChannelAttention_53343493816494
// MI455X (gfx1250) — matrix operand order repaired and checked by reading, not compiled, not run on hardware
//
#include <hip/hip_runtime.h>
#include <hip/hip_bf16.h>

// MI455X / gfx1250, wave32. Bandwidth-bound (~300MB total traffic, ~13us at
// 23.3 TB/s). Matrix work via v_wmma_f32_16x16x32_f16 with fragments loaded
// directly from global/LDS in contiguous-k operand layout, k = 16*(lane/16) + s (no staging copies).

typedef __attribute__((ext_vector_type(16))) _Float16 v16h;
typedef __attribute__((ext_vector_type(8)))  _Float16 v8h;
typedef __attribute__((ext_vector_type(8)))  float    v8f;

#define B_    8
#define N_    4096
#define DIM_  384
#define NH_   8
#define HD_   48
#define SL_   12
#define CG_   32
#define VE_   64                       // e-dim padded 48 -> 64 for K=64 GEMM2
#define SCALE_ 0.14433756729740643f    // 48^-0.5

static __device__ __forceinline__ v16h cat8(v8h lo, v8h hi) {
  return __builtin_shufflevector(lo, hi, 0,1,2,3,4,5,6,7,8,9,10,11,12,13,14,15);
}
static __device__ __forceinline__ float fget(const float4& f, int u) {
  switch (u & 3) { case 0: return f.x; case 1: return f.y; case 2: return f.z; default: return f.w; }
}

// ---------------------------------------------------------------------------
// Kernel 1: de-interleave heads, fold softmax scale into Q, cast f32 -> f16.
//   Qp/Kp : [B][NH][HD][N]   (n contiguous -> WMMA fragments load directly)
//   Vp    : [B][NH][N][VE]   (e contiguous per n-row; e=48..63 zero padded)
// 2 threads share one n-row (192 channels each); grid = 8 * 32 = 256 blocks.
// ---------------------------------------------------------------------------
__global__ void __launch_bounds__(256) pack_qkv(
    const float* __restrict__ q, const float* __restrict__ k,
    const float* __restrict__ v,
    _Float16* __restrict__ Qp, _Float16* __restrict__ Kp,
    _Float16* __restrict__ Vp) {
  const int b    = blockIdx.x >> 5;
  const int n    = ((blockIdx.x & 31) << 7) + (threadIdx.x >> 1);
  const int half = threadIdx.x & 1;                 // channel half [0,192) / [192,384)
  const float4* q4 = (const float4*)(q + ((size_t)b * N_ + n) * DIM_);
  const float4* k4 = (const float4*)(k + ((size_t)b * N_ + n) * DIM_);
  const float4* v4 = (const float4*)(v + ((size_t)b * N_ + n) * DIM_);
  const int c4beg = half * 48;
#pragma unroll 8
  for (int c4 = c4beg; c4 < c4beg + 48; ++c4) {
    float4 qv = q4[c4], kv = k4[c4], vv = v4[c4];
#pragma unroll
    for (int u = 0; u < 4; ++u) {
      const int c  = c4 * 4 + u;
      const int hh = c & 7;          // head = c % 8
      const int d  = c >> 3;         // channel-in-head
      Qp[(((size_t)b * NH_ + hh) * HD_ + d) * N_ + n] = (_Float16)(fget(qv, u) * SCALE_);
      Kp[(((size_t)b * NH_ + hh) * HD_ + d) * N_ + n] = (_Float16)fget(kv, u);
      Vp[(((size_t)b * NH_ + hh) * N_ + n) * VE_ + d] = (_Float16)fget(vv, u);
    }
  }
  // zero the e = 48..63 padding (4 heads per thread-half)
  const v8h z = {};
#pragma unroll
  for (int hh = half * 4; hh < half * 4 + 4; ++hh) {
    v8h* p = (v8h*)(Vp + (((size_t)b * NH_ + hh) * N_ + n) * VE_ + 48);
    p[0] = z; p[1] = z;
  }
}

// ---------------------------------------------------------------------------
// Kernel 2a: per (b,h): attn = Q K^T with 16-way split-K (ds_add_f32 reduce),
// masked softmax, write f16 probabilities Pg[B][NH][48][64] (K zero-padded).
// grid = 64 blocks x 512 threads (16 waves, 256-wide n slice each).
// ---------------------------------------------------------------------------
__global__ void __launch_bounds__(512) attn_softmax(
    const _Float16* __restrict__ Qp, const _Float16* __restrict__ Kp,
    const unsigned char* __restrict__ mask, _Float16* __restrict__ Pg) {
  __shared__ float attnBuf[HD_ * HD_];                 // 9 KB
  __shared__ __align__(32) _Float16 Pbuf[HD_ * VE_];   // 6 KB
  const int bb = blockIdx.x >> 3, hh = blockIdx.x & 7;
  const int tid = threadIdx.x, wave = tid >> 5, lane = tid & 31;
  const int rlo = lane & 15, hi = lane >> 4;

  for (int i = tid; i < HD_ * HD_; i += 512) attnBuf[i] = 0.f;
  __syncthreads();

  const _Float16* Qb = Qp + ((size_t)(bb * NH_ + hh) * HD_) * N_;
  const _Float16* Kb = Kp + ((size_t)(bb * NH_ + hh) * HD_) * N_;
  const int nbase = wave * 256;
  v8f acc[3][3] = {};
  for (int kc = 0; kc < 8; ++kc) {
    const int n0 = nbase + kc * 32;
    v16h a[3], bm[3];
#pragma unroll
    for (int mt = 0; mt < 3; ++mt) {   // A fragment: per-lane contiguous K chunk {hi*16..hi*16+15}
      const _Float16* r = Qb + (size_t)(mt * 16 + rlo) * N_ + n0 + hi * 16;
      a[mt] = *(const v16h*)r;
    }
#pragma unroll
    for (int nt = 0; nt < 3; ++nt) {   // B fragment: lane = column e, contiguous K
      bm[nt] = *(const v16h*)(Kb + (size_t)(nt * 16 + rlo) * N_ + n0 + hi * 16);
    }
#pragma unroll
    for (int mt = 0; mt < 3; ++mt)
#pragma unroll
      for (int nt = 0; nt < 3; ++nt)
        acc[mt][nt] = __builtin_amdgcn_wmma_f32_16x16x32_f16(
            false, a[mt], false, bm[nt], (short)0, acc[mt][nt], false, false);
  }
#pragma unroll
  for (int mt = 0; mt < 3; ++mt)
#pragma unroll
    for (int nt = 0; nt < 3; ++nt)
#pragma unroll
      for (int r = 0; r < 8; ++r)      // D layout: row = mt*16 + hi*8 + r, col = rlo
        atomicAdd(&attnBuf[(mt * 16 + hi * 8 + r) * HD_ + nt * 16 + rlo], acc[mt][nt][r]);
  __syncthreads();

  // masked softmax, one row per thread
  if (tid < HD_) {
    const int d = tid;
    const unsigned char* mrow = mask + ((size_t)hh * HD_ + d) * HD_;
    const float ninf = -__builtin_inff();
    float mx = ninf;
    for (int e = 0; e < HD_; ++e) {
      float x = mrow[e] ? ninf : attnBuf[d * HD_ + e];
      mx = fmaxf(mx, x);
    }
    float s = 0.f;
    for (int e = 0; e < HD_; ++e) {
      float x = mrow[e] ? ninf : attnBuf[d * HD_ + e];
      float p = (mx == ninf) ? 0.f : __expf(x - mx);
      attnBuf[d * HD_ + e] = p;
      s += p;
    }
    const float inv = (s > 0.f) ? 1.f / s : 0.f;
    for (int e = 0; e < HD_; ++e) Pbuf[d * VE_ + e] = (_Float16)(attnBuf[d * HD_ + e] * inv);
    for (int e = HD_; e < VE_; ++e) Pbuf[d * VE_ + e] = (_Float16)0.f;
  }
  __syncthreads();

  // spill probabilities to global (48*64 halves = 6KB per (b,h))
  if (tid < HD_ * 4) {
    const int row = tid >> 2, seg = tid & 3;
    _Float16* dst = Pg + ((size_t)(bb * NH_ + hh) * HD_ + row) * VE_ + seg * 16;
    const v8h* src = (const v8h*)&Pbuf[row * VE_ + seg * 16];
    ((v8h*)dst)[0] = src[0];
    ((v8h*)dst)[1] = src[1];
  }
}

// ---------------------------------------------------------------------------
// Kernel 2b: x = P V, scatter f16 into grouped-conv layout Xp[B][12][N][32].
// grid = (b,h) x 8 n-chunks = 512 blocks x 256 threads (wave owns 64 n).
// ---------------------------------------------------------------------------
__global__ void __launch_bounds__(256) attn_v(
    const _Float16* __restrict__ Pg, const _Float16* __restrict__ Vp,
    _Float16* __restrict__ Xp) {
  const int bh = blockIdx.x & 63, nc = blockIdx.x >> 6;
  const int bb = bh >> 3, hh = bh & 7;
  const int tid = threadIdx.x, wave = tid >> 5, lane = tid & 31;
  const int rlo = lane & 15, hi = lane >> 4;

  // P fragments from L2-hot Pg (A-layout: contiguous 16-half K chunk per lane-half)
  const _Float16* Pgb = Pg + (size_t)(bb * NH_ + hh) * HD_ * VE_;
  v16h pa[3][2];
#pragma unroll
  for (int mt = 0; mt < 3; ++mt)
#pragma unroll
    for (int kk = 0; kk < 2; ++kk) {
      const _Float16* r = Pgb + (size_t)(mt * 16 + rlo) * VE_ + kk * 32 + hi * 16;
      pa[mt][kk] = *(const v16h*)r;
    }

  const _Float16* Vb = Vp + (size_t)(bb * NH_ + hh) * N_ * VE_;
  const int nbase = nc * 512 + wave * 64;
#pragma unroll
  for (int nt = 0; nt < 4; ++nt) {
    const int n0 = nbase + nt * 16;
    const _Float16* vr = Vb + (size_t)(n0 + rlo) * VE_ + hi * 16;
    __builtin_prefetch((const void*)(vr + 64 * VE_), 0, 1);  // next wave-chunk
    v16h vb0 = *(const v16h*)vr;          // K = 0..31  (lane-half split)
    v16h vb1 = *(const v16h*)(vr + 32);   // K = 32..63 (48..63 zeros)
#pragma unroll
    for (int mt = 0; mt < 3; ++mt) {
      v8f x = {};
      x = __builtin_amdgcn_wmma_f32_16x16x32_f16(false, pa[mt][0], false, vb0, (short)0, x, false, false);
      x = __builtin_amdgcn_wmma_f32_16x16x32_f16(false, pa[mt][1], false, vb1, (short)0, x, false, false);
#pragma unroll
      for (int r = 0; r < 8; ++r) {
        const int m = mt * 16 + hi * 8 + r;
        const int c = m * 8 + hh;                       // global channel
        Xp[(((size_t)bb * SL_ + (c >> 5)) * N_ + (n0 + rlo)) * CG_ + (c & 31)] = (_Float16)x[r];
      }
    }
  }
}

// ---------------------------------------------------------------------------
// Kernel 3: grouped 1x1 conv: out[b, g*32+o, n] = sum_i W[g,o,i]*Xp[b,g,n,i]+b
// grid = (96 (b,g), 4 n-chunks) x 256 threads (wave owns 128 n).
// ---------------------------------------------------------------------------
__global__ void __launch_bounds__(256) grouped_proj(
    const _Float16* __restrict__ Xp, const float* __restrict__ Wp,
    const float* __restrict__ bp, float* __restrict__ out) {
  const int b = blockIdx.x / SL_, g = blockIdx.x % SL_;
  const int tid = threadIdx.x, wave = tid >> 5, lane = tid & 31;
  const int rlo = lane & 15, hi = lane >> 4;

  v16h aw[2];
  float bias[2][8];
#pragma unroll
  for (int mt = 0; mt < 2; ++mt) {
    const float* wr = Wp + (size_t)(g * CG_ + mt * 16 + rlo) * CG_;
#pragma unroll
    for (int i = 0; i < 16; ++i) {       // A-fragment contiguous K per lane-half
      const int kk = hi * 16 + i;
      aw[mt][i] = (_Float16)wr[kk];
    }
#pragma unroll
    for (int r = 0; r < 8; ++r) bias[mt][r] = bp[g * CG_ + mt * 16 + hi * 8 + r];
  }

  const _Float16* Xb = Xp + ((size_t)b * SL_ + g) * N_ * CG_;
  float* outB = out + ((size_t)b * DIM_ + g * CG_) * N_;
  const int nbase = blockIdx.y * 1024 + wave * 128;
#pragma unroll
  for (int nt = 0; nt < 8; ++nt) {
    const int n0 = nbase + nt * 16;
    const _Float16* xr = Xb + (size_t)(n0 + rlo) * CG_ + hi * 16;
    __builtin_prefetch((const void*)(xr + 128 * CG_), 0, 1);  // next wave-chunk
    v16h xb = *(const v16h*)xr;
#pragma unroll
    for (int mt = 0; mt < 2; ++mt) {
      v8f acc = {};
      acc = __builtin_amdgcn_wmma_f32_16x16x32_f16(false, aw[mt], false, xb, (short)0, acc, false, false);
#pragma unroll
      for (int r = 0; r < 8; ++r)
        outB[(size_t)(mt * 16 + hi * 8 + r) * N_ + n0 + rlo] = acc[r] + bias[mt][r];
    }
  }
}

// ---------------------------------------------------------------------------
extern "C" void kernel_launch(void* const* d_in, const int* in_sizes, int n_in,
                              void* d_out, int out_size, void* d_ws, size_t ws_size,
                              hipStream_t stream) {
  const float*         q    = (const float*)d_in[0];
  const float*         k    = (const float*)d_in[1];
  const float*         v    = (const float*)d_in[2];
  const float*         Wp   = (const float*)d_in[3];
  const float*         bp   = (const float*)d_in[4];
  const unsigned char* mask = (const unsigned char*)d_in[5];   // bool (1B) mask
  float* out = (float*)d_out;
  (void)in_sizes; (void)n_in; (void)out_size; (void)ws_size;

  char* ws = (char*)d_ws;
  const size_t qpB = (size_t)B_ * NH_ * HD_ * N_ * sizeof(_Float16);  // 25.2 MB
  const size_t vpB = (size_t)B_ * NH_ * N_ * VE_ * sizeof(_Float16);  // 33.6 MB
  const size_t xpB = (size_t)B_ * SL_ * N_ * CG_ * sizeof(_Float16);  // 25.2 MB
  _Float16* Qp = (_Float16*)ws;
  _Float16* Kp = (_Float16*)(ws + qpB);
  _Float16* Vp = (_Float16*)(ws + 2 * qpB);
  _Float16* Xp = (_Float16*)(ws + 2 * qpB + vpB);
  _Float16* Pg = (_Float16*)(ws + 2 * qpB + vpB + xpB);               // 0.4 MB

  hipLaunchKernelGGL(pack_qkv,     dim3(B_ * (N_ / 128)),  dim3(256), 0, stream, q, k, v, Qp, Kp, Vp);
  hipLaunchKernelGGL(attn_softmax, dim3(B_ * NH_),         dim3(512), 0, stream, Qp, Kp, mask, Pg);
  hipLaunchKernelGGL(attn_v,       dim3(B_ * NH_ * 8),     dim3(256), 0, stream, Pg, Vp, Xp);
  hipLaunchKernelGGL(grouped_proj, dim3(B_ * SL_, 4),      dim3(256), 0, stream, Xp, Wp, bp, out);
}
